// GCNSampling_317827580054
// MI455X (gfx1250) — matrix operand order repaired and checked by reading, not compiled, not run on hardware
//
#include <hip/hip_runtime.h>
#include <hip/hip_bf16.h>

typedef __attribute__((ext_vector_type(16))) _Float16 v16h;
typedef __attribute__((ext_vector_type(8)))  _Float16 v8h;
typedef __attribute__((ext_vector_type(4)))  _Float16 v4h;
typedef __attribute__((ext_vector_type(8)))  float    v8f;

#define N0 200000
#define N1 25000
#define N2 5000
#define FANOUT 16
#define IN_FEATS 500
#define NH 128          // hidden
#define NCAT 256        // 2*NH
#define NC 47           // classes
#define LDSTR1 520      // padded LDS K stride (16B aligned, bank-conflict free)
#define LDSTR2 264

// w1h: 8 col-tiles * 16 k-steps * 32 lanes * 16 f16 = 65536 elems
// w2h: 3 col-tiles *  8 k-steps * 32 lanes * 16 f16 = 12288 elems
#define W1H_ELEMS (8*16*32*16)
#define W2H_ELEMS (3*8*32*16)

// ---------------------------------------------------------------------------
// Kernel 0: repack W1 (500x128, zero-pad K->512) and W2 (256x47, zero-pad N->48)
// into a contiguous-K f16 B-operand order (matching load_A): element j of lane L at k-step s, tile c:
//   K = s*32 + (L>>4)*16 + j ,  N = c*16 + (L&15)
// so the GEMM loop loads one contiguous v16h (32B) per lane per k-step.
// ---------------------------------------------------------------------------
__global__ void prep_weights(const float* __restrict__ W1, const float* __restrict__ W2,
                             _Float16* __restrict__ w1h, _Float16* __restrict__ w2h) {
    int e = blockIdx.x * blockDim.x + threadIdx.x;
    if (e < W1H_ELEMS) {
        int j = e & 15;
        int L = (e >> 4) & 31;
        int s = (e >> 9) & 15;
        int c = e >> 13;
        int k = s * 32 + (L >> 4) * 16 + j;
        int n = c * 16 + (L & 15);
        float v = (k < IN_FEATS) ? W1[(size_t)k * NH + n] : 0.f;
        w1h[e] = (_Float16)v;
    } else if (e < W1H_ELEMS + W2H_ELEMS) {
        int e2 = e - W1H_ELEMS;
        int j = e2 & 15;
        int L = (e2 >> 4) & 31;
        int s = (e2 >> 9) & 7;
        int c = e2 >> 12;
        int k = s * 32 + (L >> 4) * 16 + j;      // k < 256 always
        int n = c * 16 + (L & 15);
        float v = (n < NC) ? W2[(size_t)k * NC + n] : 0.f;
        w2h[e2] = (_Float16)v;
    }
}

// Build the v16h A operand from LDS (contiguous K order, matching packed B):
// lane half hf: j=0..15 -> K = ks + hf*16 + j
__device__ __forceinline__ v16h load_A(const _Float16* row, int ks, int hf) {
    v8h alo = *(const v8h*)(row + ks + hf * 16);
    v8h ahi = *(const v8h*)(row + ks + hf * 16 + 8);
    v16h a;
#pragma unroll
    for (int j = 0; j < 8; ++j) { a[j] = alo[j]; a[j + 8] = ahi[j]; }
    return a;
}

// ---------------------------------------------------------------------------
// Kernel 1: per 16-row tile of layer-1 nodes:
//   gather-mean 16x16 feature rows (f32, float4 bursts) -> LDS f16 tile [16][520]
//   then 8 waves x (16x16 col tile) WMMA GEMM, +bias, store h | relu(h) as f16.
// ---------------------------------------------------------------------------
__global__ __launch_bounds__(256) void layer1(const float* __restrict__ feat,
                                              const int* __restrict__ src0,
                                              const float* __restrict__ b1,
                                              const _Float16* __restrict__ w1h,
                                              _Float16* __restrict__ h1) {
    __shared__ _Float16 m0[16][LDSTR1];
    __shared__ int sidx[16][FANOUT];
    const int tid  = threadIdx.x;
    const int tile = blockIdx.x;

    { // stage neighbor indices (one per thread: 16 rows x 16 fanout = 256)
        int r = tid >> 4, s = tid & 15;
        int g = tile * 16 + r;
        sidx[r][s] = (g < N1) ? src0[(size_t)g * FANOUT + s] : 0;
    }
    // zero the K padding region [500..519]
    for (int t = tid; t < 16 * (LDSTR1 - IN_FEATS); t += 256) {
        int r = t / (LDSTR1 - IN_FEATS);
        int c = t - r * (LDSTR1 - IN_FEATS);
        m0[r][IN_FEATS + c] = (_Float16)0.f;
    }
    __syncthreads();

    // gather + mean: 16 rows x 125 float4 chunks
    for (int t = tid; t < 16 * 125; t += 256) {
        int r = t / 125, c = t - r * 125;
        float ax = 0.f, ay = 0.f, az = 0.f, aw = 0.f;
        if (tile * 16 + r < N1) {
#pragma unroll
            for (int s = 0; s < FANOUT; ++s) {
                const float4 v = *(const float4*)(feat + (size_t)sidx[r][s] * IN_FEATS + c * 4);
                ax += v.x; ay += v.y; az += v.z; aw += v.w;
            }
        }
        const float sc = 1.f / 16.f;
        v4h h;
        h[0] = (_Float16)(ax * sc); h[1] = (_Float16)(ay * sc);
        h[2] = (_Float16)(az * sc); h[3] = (_Float16)(aw * sc);
        *(v4h*)&m0[r][c * 4] = h;
    }
    __syncthreads();

    // GEMM: wave w owns output columns [16w, 16w+16)
    const int wave = tid >> 5, lane = tid & 31;
    const int hf = lane >> 4, n = lane & 15;  // C/D: N=lane%16, rows r+8*hf ; A: M=lane%16
    v8f acc = {};
    const _Float16* wb = w1h + (size_t)wave * (16 * 32 * 16);
#pragma unroll
    for (int s = 0; s < 16; ++s) {
        v16h a = load_A(&m0[n][0], s * 32, hf);
        v16h b = *(const v16h*)(wb + (size_t)s * 512 + lane * 16);
        acc = __builtin_amdgcn_wmma_f32_16x16x32_f16(false, a, false, b,
                                                     (short)0, acc, false, false);
    }
    const int col = wave * 16 + n;
    const float bias = b1[col];
#pragma unroll
    for (int r = 0; r < 8; ++r) {
        int row = tile * 16 + r + hf * 8;
        if (row < N1) {
            float h = acc[r] + bias;
            h1[(size_t)row * NCAT + col]      = (_Float16)h;
            h1[(size_t)row * NCAT + NH + col] = (_Float16)(h > 0.f ? h : 0.f);
        }
    }
}

// ---------------------------------------------------------------------------
// Kernel 2: per 16-row tile of seed nodes: gather-mean 16x16 rows of h1cat (f16)
// into LDS [16][264], then 3 waves do the 256-K WMMA GEMM -> out[5000][47] f32.
// ---------------------------------------------------------------------------
__global__ __launch_bounds__(128) void layer2(const _Float16* __restrict__ h1,
                                              const int* __restrict__ src1,
                                              const float* __restrict__ b2,
                                              const _Float16* __restrict__ w2h,
                                              float* __restrict__ out) {
    __shared__ _Float16 m1[16][LDSTR2];
    __shared__ int sidx[16][FANOUT];
    const int tid  = threadIdx.x;
    const int tile = blockIdx.x;

    for (int t = tid; t < 256; t += 128) {
        int r = t >> 4, s = t & 15;
        int g = tile * 16 + r;
        sidx[r][s] = (g < N2) ? src1[(size_t)g * FANOUT + s] : 0;
    }
    for (int t = tid; t < 16 * (LDSTR2 - NCAT); t += 128) {  // 128 pad elems
        int r = t >> 3, c = t & 7;
        m1[r][NCAT + c] = (_Float16)0.f;
    }
    __syncthreads();

    // gather + mean: 16 rows x 32 chunks of 8 f16 (16B bursts)
    for (int t = tid; t < 16 * 32; t += 128) {
        int r = t >> 5, c = t & 31;
        float acc[8] = {0.f, 0.f, 0.f, 0.f, 0.f, 0.f, 0.f, 0.f};
        if (tile * 16 + r < N2) {
#pragma unroll
            for (int s = 0; s < FANOUT; ++s) {
                v8h v = *(const v8h*)(h1 + (size_t)sidx[r][s] * NCAT + c * 8);
#pragma unroll
                for (int j = 0; j < 8; ++j) acc[j] += (float)v[j];
            }
        }
        v8h o;
#pragma unroll
        for (int j = 0; j < 8; ++j) o[j] = (_Float16)(acc[j] * (1.f / 16.f));
        *(v8h*)&m1[r][c * 8] = o;
    }
    __syncthreads();

    const int wave = tid >> 5, lane = tid & 31;
    const int hf = lane >> 4, n = lane & 15;
    if (wave < 3) {
        v8f acc = {};
        const _Float16* wb = w2h + (size_t)wave * (8 * 32 * 16);
#pragma unroll
        for (int s = 0; s < 8; ++s) {
            v16h a = load_A(&m1[n][0], s * 32, hf);
            v16h b = *(const v16h*)(wb + (size_t)s * 512 + lane * 16);
            acc = __builtin_amdgcn_wmma_f32_16x16x32_f16(false, a, false, b,
                                                         (short)0, acc, false, false);
        }
        const int col = wave * 16 + n;
        if (col < NC) {
            const float bias = b2[col];
#pragma unroll
            for (int r = 0; r < 8; ++r) {
                int row = tile * 16 + r + hf * 8;
                if (row < N2) out[(size_t)row * NC + col] = acc[r] + bias;
            }
        }
    }
}

// ---------------------------------------------------------------------------
extern "C" void kernel_launch(void* const* d_in, const int* in_sizes, int n_in,
                              void* d_out, int out_size, void* d_ws, size_t ws_size,
                              hipStream_t stream) {
    const float* feat = (const float*)d_in[0];
    const int*   src0 = (const int*)d_in[1];
    const int*   src1 = (const int*)d_in[2];
    const float* W1   = (const float*)d_in[3];
    const float* b1   = (const float*)d_in[4];
    const float* W2   = (const float*)d_in[5];
    const float* b2   = (const float*)d_in[6];
    float* out = (float*)d_out;

    char* ws = (char*)d_ws;
    _Float16* w1h = (_Float16*)ws;                                    // 128 KB
    _Float16* w2h = (_Float16*)(ws + W1H_ELEMS * sizeof(_Float16));   //  24 KB
    _Float16* h1  = (_Float16*)(ws + (W1H_ELEMS + W2H_ELEMS) * sizeof(_Float16)); // 12.8 MB

    const int prepE = W1H_ELEMS + W2H_ELEMS;
    prep_weights<<<(prepE + 255) / 256, 256, 0, stream>>>(W1, W2, w1h, w2h);
    layer1<<<(N1 + 15) / 16, 256, 0, stream>>>(feat, src0, b1, w1h, h1);
    layer2<<<(N2 + 15) / 16, 128, 0, stream>>>(h1, src1, b2, w2h, out);
}
